// CrossAttention_70866960384676
// MI455X (gfx1250) — matrix operand order repaired and checked by reading, not compiled, not run on hardware
//
#include <hip/hip_runtime.h>

// ---------------- types for WMMA ----------------
typedef __attribute__((ext_vector_type(16))) __bf16        v16bf;
typedef __attribute__((ext_vector_type(8)))  float         v8f;
typedef __attribute__((ext_vector_type(4)))  unsigned int  uint4v;

union FragAB { uint4v u[2]; v16bf v; };

__device__ __forceinline__ unsigned short f2bf(float f) {
  unsigned int b = __float_as_uint(f);
  unsigned int r = b + 0x7FFFu + ((b >> 16) & 1u);   // round-to-nearest-even
  return (unsigned short)(r >> 16);
}

// async global->LDS copy (CDNA5, tracked by ASYNCcnt); lds_addr = 32-bit LDS byte addr
__device__ __forceinline__ void async_b128(const void* gaddr, unsigned int lds_addr) {
  asm volatile("global_load_async_to_lds_b128 %0, %1, off"
               :: "v"(lds_addr), "v"(gaddr) : "memory");
}
__device__ __forceinline__ void wait_async0() {
  asm volatile("s_wait_asynccnt 0x0" ::: "memory");
}

// ---------------- problem constants ----------------
#define BATCH 16
#define SEQ   4096
#define DIMC  1024
#define HN    16
#define ROWS  128              // keys per chunk
#define NCHUNK (SEQ / ROWS)    // 32
#define XPAD  8
#define XSTR  (DIMC + XPAD)    // ushort stride: 1032 -> 2064B pitch (16B aligned, bank-spread)
#define NKC   16               // K dim processed in 16 chunks of 2x32=64
#define LN1E4_OVER_32 0.28782313662425572f  // ln(10000)/32

// LDS layout (dynamic): xs | q_s | W double-buffer | scores | osc
#define OFF_QS   (ROWS * XSTR * 2)                 // 264192
#define OFF_BUF  (OFF_QS + 4096)                   // 268288
#define BUFSZ    16384                             // per buffer: 8KB K-tile + 8KB V-tile
#define OFF_SC   (OFF_BUF + 2 * BUFSZ)             // 301056
#define OFF_OSC  (OFF_SC + ROWS * 4)               // 301568
#define SMEM_SZ  (OFF_OSC + 8 * 64 * 4)            // 303616 B  (< 320KB WGP LDS)

// ---------------- kernel 1: q = x[:, 0, :] @ Wq, then RoPE ----------------
__global__ __launch_bounds__(256) void qproj_kernel(
    const float* __restrict__ x, const float* __restrict__ Wq,
    float* __restrict__ qrope) {
  __shared__ float xr[DIMC];
  __shared__ float qr[DIMC];
  const int b = blockIdx.x;
  for (int k = threadIdx.x; k < DIMC; k += 256)
    xr[k] = x[(size_t)b * SEQ * DIMC + k];
  __syncthreads();
  for (int j = 0; j < 4; ++j) {
    const int c = threadIdx.x + (j << 8);
    float acc = 0.f;
    for (int k = 0; k < DIMC; ++k) acc = fmaf(xr[k], Wq[k * DIMC + c], acc);
    qr[c] = acc;
  }
  __syncthreads();
  for (int j = 0; j < 4; ++j) {
    const int c = threadIdx.x + (j << 8);
    const int h = c >> 6, d = c & 63, i = d & 31;
    const float ang = (float)h * __expf(-(float)i * LN1E4_OVER_32);
    const float partner = (d < 32) ? -qr[(h << 6) + d + 32] : qr[(h << 6) + d - 32];
    qrope[b * DIMC + c] = qr[c] * __cosf(ang) + partner * __sinf(ang);
  }
}

// ------- kernel 2: Wk/Wv -> bf16, pre-swizzled into WMMA B-fragment order ------
// layout: [kb(32)][nb(64)][lane(32)][e(16)] ; k = 32*kb + 16*(lane>>4) + e ; n = 16*nb + (lane&15)
__global__ __launch_bounds__(256) void wconv_kernel(
    const float* __restrict__ Wk, const float* __restrict__ Wv,
    unsigned short* __restrict__ wk_sw, unsigned short* __restrict__ wv_sw) {
  const int idx = blockIdx.x * 256 + threadIdx.x;   // 0 .. 1M-1
  const int e = idx & 15, lane = (idx >> 4) & 31, nb = (idx >> 9) & 63, kb = idx >> 15;
  const int k = kb * 32 + ((lane >> 4) << 4) + e;
  const int n = (nb << 4) + (lane & 15);
  const float* src = (blockIdx.y == 0) ? Wk : Wv;
  unsigned short* dst = (blockIdx.y == 0) ? wk_sw : wv_sw;
  dst[idx] = f2bf(src[k * DIMC + n]);
}

// issue async loads for one (head, k-chunk) pair of 8KB K/V weight tiles
__device__ __forceinline__ void stage_async(
    const unsigned short* __restrict__ wk_sw, const unsigned short* __restrict__ wv_sw,
    int h, int kc, unsigned int buf_lds, int tid) {
  #pragma unroll
  for (int i = 0; i < 2; ++i) {
    const int vIdx = tid + (i << 8);          // 0..511 16-byte vectors per matrix
    const int blk = vIdx >> 6, within = vIdx & 63;
    const int ks = (kc << 1) + (blk >> 2), t = blk & 3;
    const size_t gb = ((size_t)((ks << 6) + (h << 2) + t) << 9) + (within << 3);
    async_b128(wk_sw + gb, buf_lds + (vIdx << 4));
    async_b128(wv_sw + gb, buf_lds + 8192u + (vIdx << 4));
  }
}

// ---------------- kernel 3: fused K/V projection + RoPE(K) + flash chunk ----------------
__global__ __launch_bounds__(256) void attn_chunk_kernel(
    const float* __restrict__ x,
    const unsigned short* __restrict__ wk_sw,
    const unsigned short* __restrict__ wv_sw,
    const float* __restrict__ qrope,
    float* __restrict__ part) {
  extern __shared__ char smem[];
  unsigned short* xs  = (unsigned short*)smem;
  float*          q_s = (float*)(smem + OFF_QS);
  float*       scores = (float*)(smem + OFF_SC);
  float*          osc = (float*)(smem + OFF_OSC);
  const unsigned int lds_base = (unsigned int)(size_t)smem;   // LDS byte address of smem base

  const int tid   = threadIdx.x;
  const int chunk = blockIdx.x;      // 0..31
  const int b     = blockIdx.y;      // 0..15
  const int j0    = chunk * ROWS;
  const int w     = tid >> 5;
  const int lane  = tid & 31;
  const int n15   = lane & 15;
  const int half  = lane >> 4;
  const int arow  = (w << 4) + n15;  // A-matrix row this lane feeds

  // stage pre-rotated, pre-scaled q
  for (int c = tid; c < DIMC; c += 256) q_s[c] = qrope[b * DIMC + c] * 0.125f;

  // stage 128x1024 activation tile as bf16 (x read from HBM exactly once)
  const size_t xbase = ((size_t)b * SEQ + j0) * DIMC;
  for (int s = tid; s < ROWS * 256; s += 256) {
    const int row = s >> 8, c4 = (s & 255) << 2;
    const float4 xv = *(const float4*)(x + xbase + (size_t)row * DIMC + c4);
    const unsigned int lo = (unsigned)f2bf(xv.x) | ((unsigned)f2bf(xv.y) << 16);
    const unsigned int hi = (unsigned)f2bf(xv.z) | ((unsigned)f2bf(xv.w) << 16);
    *(uint2*)(&xs[row * XSTR + c4]) = make_uint2(lo, hi);
  }
  __syncthreads();

  for (int h = 0; h < HN; ++h) {
    v8f accK[4] = {};
    v8f accV[4] = {};

    // prologue: prefetch k-chunk 0 into buffer 0
    stage_async(wk_sw, wv_sw, h, 0, lds_base + OFF_BUF, tid);

    for (int kc = 0; kc < NKC; ++kc) {
      wait_async0();                 // current buffer's async copies done (this wave)
      __syncthreads();               // ... and every other wave's too
      if (kc + 1 < NKC)              // prefetch next chunk into the other buffer
        stage_async(wk_sw, wv_sw, h, kc + 1,
                    lds_base + OFF_BUF + (((kc + 1) & 1) ? BUFSZ : 0u), tid);

      const char* bb = smem + OFF_BUF + ((kc & 1) ? BUFSZ : 0);
      #pragma unroll
      for (int sl = 0; sl < 2; ++sl) {
        const int k0 = ((kc << 1) + sl) << 5;
        FragAB fa;
        const unsigned short* ap = &xs[arow * XSTR + k0 + (half << 4)];
        fa.u[0] = *(const uint4v*)ap;
        fa.u[1] = *(const uint4v*)(ap + 8);
        // batch-load all 8 B fragments, then issue 8 back-to-back WMMAs
        FragAB fbk[4], fbv[4];
        #pragma unroll
        for (int t = 0; t < 4; ++t) {
          const char* pk = bb + (((sl << 2) + t) << 10) + (lane << 5);
          fbk[t].u[0] = *(const uint4v*)pk;
          fbk[t].u[1] = *(const uint4v*)(pk + 16);
          const char* pv = pk + 8192;
          fbv[t].u[0] = *(const uint4v*)pv;
          fbv[t].u[1] = *(const uint4v*)(pv + 16);
        }
        #pragma unroll
        for (int t = 0; t < 4; ++t) {
          accK[t] = __builtin_amdgcn_wmma_f32_16x16x32_bf16(
              false, fa.v, false, fbk[t].v, (short)0, accK[t], false, false);
          accV[t] = __builtin_amdgcn_wmma_f32_16x16x32_bf16(
              false, fa.v, false, fbv[t].v, (short)0, accV[t], false, false);
        }
      }
    }

    // RoPE on K fragments + scores s_j = scale * q_h . k_j   (C frag: M=rr+8*half, N=n15+16t)
    float sacc[8];
    #pragma unroll
    for (int rr = 0; rr < 8; ++rr) sacc[rr] = 0.f;
    #pragma unroll
    for (int t = 0; t < 4; ++t) {
      const int d = (t << 4) + n15, i = d & 31;
      const float ang = (float)h * __expf(-(float)i * LN1E4_OVER_32);
      const float cs = __cosf(ang), sn = __sinf(ang);
      const float sgn = (t < 2) ? -1.f : 1.f;   // rotate_half pairs d <-> d+-32 (frag t ^ 2)
      const float qv = q_s[(h << 6) + d];
      #pragma unroll
      for (int rr = 0; rr < 8; ++rr) {
        const float kr = accK[t][rr] * cs + sgn * accK[t ^ 2][rr] * sn;
        sacc[rr] += qv * kr;
      }
    }
    #pragma unroll
    for (int rr = 0; rr < 8; ++rr) {
      #pragma unroll
      for (int off = 1; off < 16; off <<= 1)
        sacc[rr] += __shfl_xor(sacc[rr], off, 32);
    }
    if (n15 == 0) {
      #pragma unroll
      for (int rr = 0; rr < 8; ++rr)
        scores[(w << 4) + (half << 3) + rr] = sacc[rr];
    }
    __syncthreads();

    // chunk-local softmax partials + O = sum_j p_j * V_j
    float mx = -1e30f;
    for (int j = 0; j < ROWS; ++j) mx = fmaxf(mx, scores[j]);
    float p[8];
    #pragma unroll
    for (int rr = 0; rr < 8; ++rr)
      p[rr] = __expf(scores[(w << 4) + (half << 3) + rr] - mx);
    #pragma unroll
    for (int t = 0; t < 4; ++t) {
      float o = 0.f;
      #pragma unroll
      for (int rr = 0; rr < 8; ++rr) o += p[rr] * accV[t][rr];
      o += __shfl_xor(o, 16, 32);              // fold both row-halves
      if (half == 0) osc[(w << 6) + (t << 4) + n15] = o;
    }
    __syncthreads();

    float* pb = &part[(size_t)(((b * NCHUNK + chunk) * HN) + h) * 68];
    if (tid < 64) {
      float oo = 0.f;
      for (int ww = 0; ww < 8; ++ww) oo += osc[(ww << 6) + tid];
      pb[tid] = oo;
    }
    if (tid == 0) {
      float l = 0.f;
      for (int j = 0; j < ROWS; ++j) l += __expf(scores[j] - mx);
      pb[64] = mx;
      pb[65] = l;
    }
    __syncthreads();   // scores/osc reused by next head
  }
}

// ---------------- kernel 4: flash combine across chunks ----------------
__global__ __launch_bounds__(64) void combine_kernel(
    const float* __restrict__ part, float* __restrict__ attn_out) {
  const int bh = blockIdx.x;
  const int b = bh >> 4, h = bh & 15;
  const int d = threadIdx.x;
  float M = -1e30f;
  for (int c = 0; c < NCHUNK; ++c)
    M = fmaxf(M, part[(size_t)(((b * NCHUNK + c) * HN) + h) * 68 + 64]);
  float L = 0.f, acc = 0.f;
  for (int c = 0; c < NCHUNK; ++c) {
    const float* p = &part[(size_t)(((b * NCHUNK + c) * HN) + h) * 68];
    const float wgt = __expf(p[64] - M);
    L += wgt * p[65];
    acc += wgt * p[d];
  }
  attn_out[b * DIMC + (h << 6) + d] = acc / L;
}

// ---------------- kernel 5: out = attn_out @ Wp + bp ----------------
__global__ __launch_bounds__(256) void outproj_kernel(
    const float* __restrict__ attn, const float* __restrict__ Wp,
    const float* __restrict__ bp, float* __restrict__ out) {
  __shared__ float rowv[DIMC];
  const int b = blockIdx.x >> 2;
  const int c = ((blockIdx.x & 3) << 8) + threadIdx.x;
  for (int k = threadIdx.x; k < DIMC; k += 256) rowv[k] = attn[b * DIMC + k];
  __syncthreads();
  float acc = bp[c];
  for (int k = 0; k < DIMC; ++k) acc = fmaf(rowv[k], Wp[k * DIMC + c], acc);
  out[b * DIMC + c] = acc;
}

// ---------------- launcher ----------------
extern "C" void kernel_launch(void* const* d_in, const int* in_sizes, int n_in,
                              void* d_out, int out_size, void* d_ws, size_t ws_size,
                              hipStream_t stream) {
  const float* x  = (const float*)d_in[0];
  const float* Wq = (const float*)d_in[1];
  const float* Wk = (const float*)d_in[2];
  const float* Wv = (const float*)d_in[3];
  const float* Wp = (const float*)d_in[4];
  const float* bp = (const float*)d_in[5];
  float* out = (float*)d_out;

  char* ws = (char*)d_ws;
  float*          q_rope   = (float*)(ws);                               // 64 KB
  unsigned short* wk_sw    = (unsigned short*)(ws + 65536);              // 2 MB
  unsigned short* wv_sw    = (unsigned short*)(ws + 65536 + 2097152);    // 2 MB
  float*          part     = (float*)(ws + 65536 + 2 * 2097152);         // 2.23 MB
  float*          attn_out = (float*)(ws + 65536 + 2 * 2097152 + 2228224);

  qproj_kernel<<<BATCH, 256, 0, stream>>>(x, Wq, q_rope);
  wconv_kernel<<<dim3(4096, 2), 256, 0, stream>>>(Wk, Wv, wk_sw, wv_sw);
  attn_chunk_kernel<<<dim3(NCHUNK, BATCH), 256, SMEM_SZ, stream>>>(
      x, wk_sw, wv_sw, q_rope, part);
  combine_kernel<<<BATCH * HN, 64, 0, stream>>>(part, attn_out);
  outproj_kernel<<<BATCH * 4, 256, 0, stream>>>(attn_out, Wp, bp, out);
}
